// DbrxAttention_36627481100904
// MI455X (gfx1250) — matrix operand order repaired and checked by reading, not compiled, not run on hardware
//
#include <hip/hip_runtime.h>
#include <cstdint>
#include <cstddef>

// ---------------------------------------------------------------------------
// Types for CDNA5 WMMA (wave32): bf16 A/B fragments (16 x bf16 = 8 VGPRs),
// fp32 C/D accumulators (8 x f32 = 8 VGPRs).
// ---------------------------------------------------------------------------
typedef float  v8f   __attribute__((ext_vector_type(8)));
typedef float  f32x4 __attribute__((ext_vector_type(4)));
typedef __bf16 v16bf __attribute__((ext_vector_type(16)));

union Frag16 {
  v16bf bf;
  f32x4 q[2];   // two 16-byte halves (VGPR0-3, VGPR4-7)
};

static __device__ __forceinline__ v8f wmma_bf16(Frag16 a, Frag16 b, v8f c) {
  // D = A(16x32 bf16) * B(32x16 bf16) + C(16x16 f32)
  return __builtin_amdgcn_wmma_f32_16x16x32_bf16(
      /*neg_a=*/false, a.bf, /*neg_b=*/false, b.bf,
      /*c_mod=*/(short)0, c, /*reuse_a=*/false, /*reuse_b=*/false);
}

// ---------------------------------------------------------------------------
// Direct async global->LDS copies (GLOBAL_LOAD_ASYNC_TO_LDS_B128, ASYNCcnt)
// when the toolchain exposes the builtin; otherwise VGPR-bounce fallback.
// Builtin signature (from hipcc diagnostic): (int4 AS1*, int4 AS3*, Ii, Ii).
// ---------------------------------------------------------------------------
#ifdef __has_builtin
#if __has_builtin(__builtin_amdgcn_global_load_async_to_lds_b128)
#define HAVE_ASYNC_LDS 1
#endif
#endif
#ifndef HAVE_ASYNC_LDS
#define HAVE_ASYNC_LDS 0
#endif

#if HAVE_ASYNC_LDS
typedef int v4i_t __attribute__((ext_vector_type(4)));
typedef __attribute__((address_space(1))) v4i_t* gptr_b128;
typedef __attribute__((address_space(3))) v4i_t* lptr_b128;
#endif

static __device__ __forceinline__ void copy16_to_lds(const void* g, void* l) {
#if HAVE_ASYNC_LDS
  // LDS generic addresses truncate to the in-wave LDS offset (ISA 10.2).
  __builtin_amdgcn_global_load_async_to_lds_b128(
      (gptr_b128)(unsigned long long)(uintptr_t)g,
      (lptr_b128)(unsigned int)(uintptr_t)l, 0, 0);
#else
  *(f32x4*)l = *(const f32x4*)g;
#endif
}

static __device__ __forceinline__ void lds_copy_fence() {
#if HAVE_ASYNC_LDS
  asm volatile("s_wait_asynccnt 0" ::: "memory");  // __syncthreads only waits DScnt
#endif
}

#define S_LEN   2048
#define D_MODEL 4096
#define NHQ     32
#define NKVH    8
#define HDIM    128
#define CLIPV   8.0f
#define SCALEV  0.08838834764831845f   // 128^-0.5

// ---------------------------------------------------------------------------
// Kernel 0: fp32 -> bf16 conversion, 8 elements/thread (grid-stride)
// ---------------------------------------------------------------------------
__global__ __launch_bounds__(256) void cvt_f32_bf16(const float* __restrict__ x,
                                                    __bf16* __restrict__ y, int n) {
  union Pack8 { __bf16 h[8]; f32x4 v; };
  int i = (blockIdx.x * blockDim.x + threadIdx.x) * 8;
  const int stride = gridDim.x * blockDim.x * 8;
  for (; i < n; i += stride) {
    const f32x4 a = *(const f32x4*)(x + i);
    const f32x4 b = *(const f32x4*)(x + i + 4);
    Pack8 p;
#pragma unroll
    for (int j = 0; j < 4; ++j) { p.h[j] = (__bf16)a[j]; p.h[j + 4] = (__bf16)b[j]; }
    *(f32x4*)(y + i) = p.v;
  }
}

// ---------------------------------------------------------------------------
// Kernel 1: QKV projection GEMM (M=B*S=4096, N=6144, K=4096) + clamp + RoPE.
// 128x128 block tile; blockIdx.y == one head (128 cols) so RoPE rotate-half
// pairing (d, d+64) is wave-local. Wave w: rows [16w,16w+16) x 128 cols.
// Double-buffered LDS staging; 8 B-fragments register-blocked per K-step so
// the 8 WMMAs issue back-to-back behind a single DS wait.
// ---------------------------------------------------------------------------
__global__ __launch_bounds__(256) void qkv_gemm_kernel(
    const __bf16* __restrict__ A,     // [4096, 4096] hidden bf16
    const __bf16* __restrict__ W,     // [6144, 4096] Wqkv bf16 (row n = out col)
    const float*  __restrict__ cosT,  // [2048, 128]
    const float*  __restrict__ sinT,  // [2048, 128]
    __bf16* __restrict__ Qb,          // [B,NH,S,HD]  (pre-scaled by HD^-0.5)
    __bf16* __restrict__ Kb,          // [B,NKV,S,HD]
    __bf16* __restrict__ Vt)          // [B,NKV,HD,S] (transposed for PV WMMA)
{
  __shared__ __align__(16) __bf16 As[2][128][40];  // 32 + 8 pad
  __shared__ __align__(16) __bf16 Bs[2][128][40];

  const int tid  = threadIdx.x;
  const int wave = tid >> 5;
  const int lane = tid & 31;
  const int m    = lane & 15;     // row (A) / col (B,C) within 16
  const int half = lane >> 4;     // K-half selector per ISA fragment layout
  const int bm   = blockIdx.x;    // M tile (32 of them)
  const int head = blockIdx.y;    // 0..47: 0-31 Q, 32-39 K, 40-47 V

  const int lr = tid >> 1;               // 0..127
  const int lc = (tid & 1) * 16;         // 0 or 16
  const __bf16* gA = A + (size_t)(bm * 128 + lr) * 4096 + lc;
  const __bf16* gB = W + (size_t)(head * 128 + lr) * 4096 + lc;

  v8f acc[8] = {};

  auto stage = [&](int p, int k0) {
    copy16_to_lds(gA + k0,     &As[p][lr][lc]);
    copy16_to_lds(gA + k0 + 8, &As[p][lr][lc + 8]);
    copy16_to_lds(gB + k0,     &Bs[p][lr][lc]);
    copy16_to_lds(gB + k0 + 8, &Bs[p][lr][lc + 8]);
    if (k0 + 32 < 4096) {
      __builtin_prefetch(gA + k0 + 32, 0, 1);   // global_prefetch_b8
      __builtin_prefetch(gB + k0 + 32, 0, 1);
    }
  };

  stage(0, 0);
  for (int kt = 0; kt < 128; ++kt) {
    const int cur = kt & 1;
    lds_copy_fence();
    __syncthreads();                       // tile `cur` visible to all waves
    if (kt + 1 < 128) stage(cur ^ 1, (kt + 1) * 32);

    // A fragment (16-bit A 16x32): lanes 0-15 K=0..7 & 16..23; lanes 16-31 +8
    Frag16 a;
    a.q[0] = *(const f32x4*)&As[cur][wave * 16 + m][8 * half];
    a.q[1] = *(const f32x4*)&As[cur][wave * 16 + m][16 + 8 * half];
    // Register-block all 8 B fragments, then issue 8 WMMAs back-to-back.
    Frag16 bfr[8];
#pragma unroll
    for (int t = 0; t < 8; ++t) {
      bfr[t].q[0] = *(const f32x4*)&Bs[cur][t * 16 + m][8 * half];
      bfr[t].q[1] = *(const f32x4*)&Bs[cur][t * 16 + m][16 + 8 * half];
    }
#pragma unroll
    for (int t = 0; t < 8; ++t) acc[t] = wmma_bf16(a, bfr[t], acc[t]);
  }

  // Epilogue: C layout row = e + 8*half, col = t*16 + m
#pragma unroll
  for (int e = 0; e < 8; ++e) {
    const int mrow = bm * 128 + wave * 16 + e + 8 * half;   // 0..4095 over B*S
    const int b = mrow >> 11;
    const int s = mrow & 2047;
    float vals[8];
#pragma unroll
    for (int t = 0; t < 8; ++t) {
      float v = acc[t][e];
      vals[t] = fminf(fmaxf(v, -CLIPV), CLIPV);
    }
    if (head < 40) {   // RoPE on Q and K heads: out[d]=x[d]*c[d]+rot(x)[d]*s[d]
#pragma unroll
      for (int t = 0; t < 4; ++t) {
        const int d1 = t * 16 + m;
        const int d2 = d1 + 64;
        const float c1 = cosT[s * 128 + d1], s1 = sinT[s * 128 + d1];
        const float c2 = cosT[s * 128 + d2], s2 = sinT[s * 128 + d2];
        const float x1 = vals[t], x2 = vals[t + 4];
        vals[t]     = x1 * c1 - x2 * s1;
        vals[t + 4] = x2 * c2 + x1 * s2;
      }
    }
    if (head < 32) {
      __bf16* dst = Qb + ((size_t)(b * NHQ + head) * S_LEN + s) * HDIM;
#pragma unroll
      for (int t = 0; t < 8; ++t) dst[t * 16 + m] = (__bf16)(vals[t] * SCALEV);
    } else if (head < 40) {
      __bf16* dst = Kb + ((size_t)(b * NKVH + (head - 32)) * S_LEN + s) * HDIM;
#pragma unroll
      for (int t = 0; t < 8; ++t) dst[t * 16 + m] = (__bf16)vals[t];
    } else {
      __bf16* dst = Vt + (size_t)(b * NKVH + (head - 40)) * HDIM * S_LEN;
#pragma unroll
      for (int t = 0; t < 8; ++t)
        dst[(size_t)(t * 16 + m) * S_LEN + s] = (__bf16)vals[t];
    }
  }
}

// ---------------------------------------------------------------------------
// Kernel 2: causal flash attention, GQA (4 q-heads per kv-head).
// grid = (S/128, B*NH). 8 waves x 16 q-rows = 128 q-rows per block.
// Double-buffered K/V^T staging; fragments register-blocked in groups of 4.
// ---------------------------------------------------------------------------
__global__ __launch_bounds__(256) void flash_attn_kernel(
    const __bf16* __restrict__ Qb, const __bf16* __restrict__ Kb,
    const __bf16* __restrict__ Vt, __bf16* __restrict__ Ob)
{
  __shared__ __align__(16) __bf16 Ks[2][32][136];   // K tile: 32 x 128 d (+8 pad)
  __shared__ __align__(16) __bf16 Vs[2][128][40];   // V^T tile: 128 d x 32 k (+8)
  __shared__ __align__(16) __bf16 Ps[8][16][40];    // per-wave P relayout buffer

  const int tid  = threadIdx.x;
  const int wave = tid >> 5;
  const int lane = tid & 31;
  const int m    = lane & 15;
  const int half = lane >> 4;
  const int bh = blockIdx.y;
  const int b  = bh >> 5;
  const int h  = bh & 31;
  const int kv = h >> 2;            // NH/NKV = 4
  const int qb = blockIdx.x * 128;

  // Q fragments held in registers: 4 d-chunks of 32
  const __bf16* qrow =
      Qb + ((size_t)(b * NHQ + h) * S_LEN + qb + wave * 16 + m) * HDIM;
  Frag16 qf[4];
#pragma unroll
  for (int d = 0; d < 4; ++d) {
    qf[d].q[0] = *(const f32x4*)(qrow + d * 32 + 8 * half);
    qf[d].q[1] = *(const f32x4*)(qrow + d * 32 + 16 + 8 * half);
  }

  const __bf16* Kbase = Kb + (size_t)(b * NKVH + kv) * S_LEN * HDIM;
  const __bf16* Vbase = Vt + (size_t)(b * NKVH + kv) * HDIM * S_LEN;

  v8f o[8] = {};
  float mi[8], li[8];
#pragma unroll
  for (int e = 0; e < 8; ++e) { mi[e] = -1e30f; li[e] = 0.0f; }

  const int nkt = blockIdx.x * 4 + 4;    // k tiles of 32 up to q-block end
  const int sKr = tid >> 3, sKc = (tid & 7) * 16;
  const int sVd = tid >> 1, sVc = (tid & 1) * 16;

  auto stage = [&](int p, int kb0) {
    const __bf16* gK = Kbase + (size_t)(kb0 + sKr) * HDIM + sKc;
    copy16_to_lds(gK,     &Ks[p][sKr][sKc]);
    copy16_to_lds(gK + 8, &Ks[p][sKr][sKc + 8]);
    const __bf16* gV = Vbase + (size_t)sVd * S_LEN + kb0 + sVc;
    copy16_to_lds(gV,     &Vs[p][sVd][sVc]);
    copy16_to_lds(gV + 8, &Vs[p][sVd][sVc + 8]);
    __builtin_prefetch(gK + 32 * HDIM, 0, 1);
  };

  stage(0, 0);
  for (int kt = 0; kt < nkt; ++kt) {
    const int cur = kt & 1;
    const int kb0 = kt * 32;
    lds_copy_fence();
    __syncthreads();
    if (kt + 1 < nkt) stage(cur ^ 1, kb0 + 32);

    // S = Q K^T : two 16-col score tiles; 4 K-fragments blocked, then 4 WMMAs
    v8f sacc[2] = {};
#pragma unroll
    for (int j = 0; j < 2; ++j) {
      Frag16 kf[4];
#pragma unroll
      for (int d = 0; d < 4; ++d) {
        kf[d].q[0] = *(const f32x4*)&Ks[cur][j * 16 + m][d * 32 + 8 * half];
        kf[d].q[1] = *(const f32x4*)&Ks[cur][j * 16 + m][d * 32 + 16 + 8 * half];
      }
#pragma unroll
      for (int d = 0; d < 4; ++d) sacc[j] = wmma_bf16(qf[d], kf[d], sacc[j]);
    }

    // Causal mask + online softmax. Row of vgpr e / lane-group = e + 8*half.
#pragma unroll
    for (int e = 0; e < 8; ++e) {
      const int qrow_i = qb + wave * 16 + e + 8 * half;
      float mx = -1e30f;
#pragma unroll
      for (int j = 0; j < 2; ++j) {
        const int col = kb0 + j * 16 + m;
        float v = sacc[j][e];
        if (col > qrow_i) v = -1e30f;
        sacc[j][e] = v;
        mx = fmaxf(mx, v);
      }
#pragma unroll
      for (int off = 1; off < 16; off <<= 1)     // 16-lane butterfly (wave32)
        mx = fmaxf(mx, __shfl_xor(mx, off, 32));
      const float mnew  = fmaxf(mi[e], mx);
      const float alpha = __expf(mi[e] - mnew);
      float rsum = 0.0f;
#pragma unroll
      for (int j = 0; j < 2; ++j) {
        const float p = __expf(sacc[j][e] - mnew);
        sacc[j][e] = p;
        rsum += p;
      }
#pragma unroll
      for (int off = 1; off < 16; off <<= 1)
        rsum += __shfl_xor(rsum, off, 32);
      li[e] = li[e] * alpha + rsum;
      mi[e] = mnew;
#pragma unroll
      for (int t = 0; t < 8; ++t) o[t][e] *= alpha;
    }

    // Re-layout P (C layout) -> A fragment via per-wave LDS buffer
#pragma unroll
    for (int e = 0; e < 8; ++e)
#pragma unroll
      for (int j = 0; j < 2; ++j)
        Ps[wave][e + 8 * half][j * 16 + m] = (__bf16)sacc[j][e];
    asm volatile("s_wait_dscnt 0" ::: "memory");  // wave-local LDS RAW fence
    Frag16 pf;
    pf.q[0] = *(const f32x4*)&Ps[wave][m][8 * half];
    pf.q[1] = *(const f32x4*)&Ps[wave][m][16 + 8 * half];

    // O += P V : 8 d-column tiles; V-fragments blocked in groups of 4
#pragma unroll
    for (int g = 0; g < 2; ++g) {
      Frag16 vf[4];
#pragma unroll
      for (int t = 0; t < 4; ++t) {
        const int dr = (g * 4 + t) * 16 + m;
        vf[t].q[0] = *(const f32x4*)&Vs[cur][dr][8 * half];
        vf[t].q[1] = *(const f32x4*)&Vs[cur][dr][16 + 8 * half];
      }
#pragma unroll
      for (int t = 0; t < 4; ++t)
        o[g * 4 + t] = wmma_bf16(pf, vf[t], o[g * 4 + t]);
    }
  }

  // Normalize and store O as bf16 in [B, S, NH*HD] for the output GEMM
#pragma unroll
  for (int e = 0; e < 8; ++e) {
    const float inv = 1.0f / li[e];
    const int s = qb + wave * 16 + e + 8 * half;
    __bf16* dst = Ob + (size_t)(b * S_LEN + s) * D_MODEL + h * HDIM;
#pragma unroll
    for (int t = 0; t < 8; ++t) dst[t * 16 + m] = (__bf16)(o[t][e] * inv);
  }
}

// ---------------------------------------------------------------------------
// Kernel 3: output projection GEMM (4096 x 4096 x 4096), fp32 result.
// ---------------------------------------------------------------------------
__global__ __launch_bounds__(256) void out_gemm_kernel(
    const __bf16* __restrict__ A,    // [4096,4096] attention output bf16
    const __bf16* __restrict__ W,    // [4096,4096] Wout bf16
    float* __restrict__ out)         // [4096,4096] fp32
{
  __shared__ __align__(16) __bf16 As[2][128][40];
  __shared__ __align__(16) __bf16 Bs[2][128][40];

  const int tid  = threadIdx.x;
  const int wave = tid >> 5;
  const int lane = tid & 31;
  const int m    = lane & 15;
  const int half = lane >> 4;
  const int bm = blockIdx.x;
  const int bn = blockIdx.y;

  const int lr = tid >> 1;
  const int lc = (tid & 1) * 16;
  const __bf16* gA = A + (size_t)(bm * 128 + lr) * 4096 + lc;
  const __bf16* gB = W + (size_t)(bn * 128 + lr) * 4096 + lc;

  v8f acc[8] = {};

  auto stage = [&](int p, int k0) {
    copy16_to_lds(gA + k0,     &As[p][lr][lc]);
    copy16_to_lds(gA + k0 + 8, &As[p][lr][lc + 8]);
    copy16_to_lds(gB + k0,     &Bs[p][lr][lc]);
    copy16_to_lds(gB + k0 + 8, &Bs[p][lr][lc + 8]);
    if (k0 + 32 < 4096) {
      __builtin_prefetch(gA + k0 + 32, 0, 1);
      __builtin_prefetch(gB + k0 + 32, 0, 1);
    }
  };

  stage(0, 0);
  for (int kt = 0; kt < 128; ++kt) {
    const int cur = kt & 1;
    lds_copy_fence();
    __syncthreads();
    if (kt + 1 < 128) stage(cur ^ 1, (kt + 1) * 32);

    Frag16 a;
    a.q[0] = *(const f32x4*)&As[cur][wave * 16 + m][8 * half];
    a.q[1] = *(const f32x4*)&As[cur][wave * 16 + m][16 + 8 * half];
    Frag16 bfr[8];
#pragma unroll
    for (int t = 0; t < 8; ++t) {
      bfr[t].q[0] = *(const f32x4*)&Bs[cur][t * 16 + m][8 * half];
      bfr[t].q[1] = *(const f32x4*)&Bs[cur][t * 16 + m][16 + 8 * half];
    }
#pragma unroll
    for (int t = 0; t < 8; ++t) acc[t] = wmma_bf16(a, bfr[t], acc[t]);
  }

#pragma unroll
  for (int e = 0; e < 8; ++e) {
    const int mrow = bm * 128 + wave * 16 + e + 8 * half;
    float* dst = out + (size_t)mrow * 4096 + bn * 128;
#pragma unroll
    for (int t = 0; t < 8; ++t) dst[t * 16 + m] = acc[t][e];
  }
}

// ---------------------------------------------------------------------------
// Launcher
// ---------------------------------------------------------------------------
extern "C" void kernel_launch(void* const* d_in, const int* in_sizes, int n_in,
                              void* d_out, int out_size, void* d_ws, size_t ws_size,
                              hipStream_t stream) {
  const float* hs   = (const float*)d_in[0];   // [2,2048,4096]
  const float* Wqkv = (const float*)d_in[1];   // [6144,4096]
  const float* Wout = (const float*)d_in[2];   // [4096,4096]
  const float* cosT = (const float*)d_in[3];   // [2048,128]
  const float* sinT = (const float*)d_in[4];   // [2048,128]
  float* out = (float*)d_out;                  // [2,2048,4096] fp32

  char* w = (char*)d_ws;
  size_t off = 0;
  auto take = [&](size_t bytes) {
    void* p = w + off;
    off += (bytes + 255) & ~(size_t)255;
    return p;
  };
  __bf16* hb    = (__bf16*)take((size_t)4096 * 4096 * 2);              // 32 MB
  __bf16* wqkvb = (__bf16*)take((size_t)6144 * 4096 * 2);              // 48 MB
  __bf16* woutb = (__bf16*)take((size_t)4096 * 4096 * 2);              // 32 MB
  __bf16* Qb    = (__bf16*)take((size_t)2 * NHQ  * S_LEN * HDIM * 2);  // 32 MB
  __bf16* Kb    = (__bf16*)take((size_t)2 * NKVH * S_LEN * HDIM * 2);  // 8 MB
  __bf16* Vtb   = (__bf16*)take((size_t)2 * NKVH * HDIM * S_LEN * 2);  // 8 MB
  __bf16* Ob    = (__bf16*)take((size_t)2 * S_LEN * D_MODEL * 2);      // 32 MB

  cvt_f32_bf16<<<2048, 256, 0, stream>>>(hs,   hb,    4096 * 4096);
  cvt_f32_bf16<<<2048, 256, 0, stream>>>(Wqkv, wqkvb, 6144 * 4096);
  cvt_f32_bf16<<<2048, 256, 0, stream>>>(Wout, woutb, 4096 * 4096);

  qkv_gemm_kernel<<<dim3(32, 48), 256, 0, stream>>>(hb, wqkvb, cosT, sinT,
                                                    Qb, Kb, Vtb);
  flash_attn_kernel<<<dim3(16, 64), 256, 0, stream>>>(Qb, Kb, Vtb, Ob);
  out_gemm_kernel<<<dim3(32, 32), 256, 0, stream>>>(Ob, woutb, out);
}
